// LPAM_head_34041910788830
// MI455X (gfx1250) — matrix operand order repaired and checked by reading, not compiled, not run on hardware
//
#include <hip/hip_runtime.h>
#include <hip/hip_bf16.h>
#include <math.h>

typedef __attribute__((ext_vector_type(16))) _Float16 v16h;
typedef __attribute__((ext_vector_type(8)))  float    v8f;
typedef __attribute__((ext_vector_type(4)))  float    f4v;   // raw vector for nontemporal builtins

#define Bn 8
#define Cn 48
#define CRn 12
#define Hn 256
#define Wn 256
#define HW (Hn*Wn)          // 65536
#define HW1 (HW/4)          // 16384  (128x128)
#define HW2 (HW/16)         // 4096   (64x64)
#define HW3 (HW/64)         // 1024   (32x32)
#define NTILES (HW/16)      // 4096 pixel tiles of 16 per batch

// h intermediate: tile-major, padded to 16 channels.
// h16t[((b*NTILES + t)*16 + k)*16 + n]  -> one 512B contiguous block per tile.

// ---------------------------------------------------------------------------
// conv1: h = relu(W1 @ x)  (per batch GEMM M=12->16, K=48->64, N=65536)
// 2 pixel tiles per iteration, branch-free padding, f16 tile-major output,
// plus per-channel sums for the SE mean.
// ---------------------------------------------------------------------------
__global__ __launch_bounds__(256) void conv1_wmma(
    const float* __restrict__ x, const float* __restrict__ w1,
    _Float16* __restrict__ h16t, float* __restrict__ hsum)
{
  const int b    = blockIdx.y;
  const int lane = threadIdx.x & 31;
  const int wave = threadIdx.x >> 5;
  const int m    = lane & 15;     // A row / D col
  const int hs   = lane >> 4;     // half-select
  // A tiles 16x32 f16: chunk0 K=0..31, chunk1 K=32..63 (>=48 zero)
  v16h a0{}, a1{};
  #pragma unroll
  for (int i = 0; i < 16; ++i) {
    int kl = i + 16*hs;
    a0[i] = (_Float16)((m < CRn) ? w1[m*Cn + kl] : 0.0f);
    a1[i] = (_Float16)((m < CRn && kl < 16) ? w1[m*Cn + 32 + kl] : 0.0f);
  }
  const int n  = lane & 15;        // B pixel column
  const int kh = lane >> 4;        // B lane-group (K/16)
  const float mask1 = (kh == 0) ? 1.0f : 0.0f;   // chunk1 valid only for kh==0
  const size_t xb = (size_t)b * Cn * HW;
  float acc[8] = {0,0,0,0,0,0,0,0};

  for (int tp = blockIdx.x*8 + wave; tp < NTILES/2; tp += gridDim.x*8) {
    const int p0 = tp * 32;                      // two 16-pixel tiles
    // ---- issue all loads first (branch-free addresses) ----
    float f0a[16], f0b[16], f1a[16], f1b[16];
    #pragma unroll
    for (int i = 0; i < 16; ++i) {
      const float* pc = x + xb + (size_t)(kh*16 + i)*HW + p0 + n;  // ch 0..31
      f0a[i] = pc[0];
      f0b[i] = pc[16];
    }
    #pragma unroll
    for (int i = 0; i < 16; ++i) {
      const float* pc = x + xb + (size_t)(32 + i)*HW + p0 + n;     // ch 32..47
      f1a[i] = pc[0];
      f1b[i] = pc[16];
    }
    // ---- convert ----
    v16h b0a{}, b0b{}, b1a{}, b1b{};
    #pragma unroll
    for (int i = 0; i < 16; ++i) {
      b0a[i] = (_Float16)f0a[i];
      b0b[i] = (_Float16)f0b[i];
      b1a[i] = (_Float16)(f1a[i] * mask1);
      b1b[i] = (_Float16)(f1b[i] * mask1);
    }
    v8f ca = {0,0,0,0,0,0,0,0}, cb = {0,0,0,0,0,0,0,0};
    ca = __builtin_amdgcn_wmma_f32_16x16x32_f16(false, a0, false, b0a, (short)0, ca, false, false);
    ca = __builtin_amdgcn_wmma_f32_16x16x32_f16(false, a1, false, b1a, (short)0, ca, false, false);
    cb = __builtin_amdgcn_wmma_f32_16x16x32_f16(false, a0, false, b0b, (short)0, cb, false, false);
    cb = __builtin_amdgcn_wmma_f32_16x16x32_f16(false, a1, false, b1b, (short)0, cb, false, false);
    // ---- relu + store tile-major (unconditional; rows 12..15 are exact 0) ----
    _Float16* ta = h16t + ((size_t)b*NTILES + 2*tp    )*256;
    _Float16* tb = h16t + ((size_t)b*NTILES + 2*tp + 1)*256;
    #pragma unroll
    for (int v = 0; v < 8; ++v) {
      int row = v + 8*hs;
      float ra = ca[v] > 0.0f ? ca[v] : 0.0f;
      float rb = cb[v] > 0.0f ? cb[v] : 0.0f;
      ta[row*16 + n] = (_Float16)ra;
      tb[row*16 + n] = (_Float16)rb;
      acc[v] += ra + rb;
    }
  }
  #pragma unroll
  for (int v = 0; v < 8; ++v) {
    int row = v + 8*hs;
    if (row < CRn) atomicAdd(&hsum[b*CRn + row], acc[v]);
  }
}

// ---------------------------------------------------------------------------
// SE gate (tiny)
// ---------------------------------------------------------------------------
__global__ void se_gate(const float* __restrict__ hsum, const float* __restrict__ kvw,
                        const float* __restrict__ kvb, float* __restrict__ gfold)
{
  int t = threadIdx.x;
  if (t >= Bn*CRn) return;
  int b = t / CRn, i = t % CRn;
  float q[CRn];
  for (int c = 0; c < CRn; ++c) q[c] = hsum[b*CRn + c] * (1.0f/(float)HW);
  const float scale = rsqrtf((float)CRn);
  float k[CRn], v[CRn];
  for (int j = 0; j < CRn; ++j) {
    float kk = kvb[j], vv = kvb[CRn + j];
    for (int c = 0; c < CRn; ++c) {
      kk += q[c]*kvw[j*CRn + c];
      vv += q[c]*kvw[(CRn + j)*CRn + c];
    }
    k[j] = kk; v[j] = vv;
  }
  float qi = q[i], mx = -1e30f, e[CRn];
  for (int j = 0; j < CRn; ++j) { e[j] = qi*k[j]*scale; mx = fmaxf(mx, e[j]); }
  float den = 0.f, num = 0.f;
  for (int j = 0; j < CRn; ++j) { float ex = __expf(e[j]-mx); den += ex; num += ex*v[j]; }
  float g = 1.0f/(1.0f + __expf(-(num/den)));
  gfold[b*CRn + i] = g + 1.0f;
}

// ---------------------------------------------------------------------------
// conv2: x1 = (W2 * diag(gate+1)) @ h  (M=48, K=12->32, N=65536), WMMA,
// 2 tiles/iter, tile-major f16 input, branch-free.
// ---------------------------------------------------------------------------
__global__ __launch_bounds__(256) void conv2_wmma(
    const _Float16* __restrict__ h16t, const float* __restrict__ w2,
    const float* __restrict__ gfold, float* __restrict__ x1)
{
  const int b    = blockIdx.y;
  const int lane = threadIdx.x & 31;
  const int wave = threadIdx.x >> 5;
  const int mrow = lane & 15;
  const int hs   = lane >> 4;
  v16h A0{}, A1{}, A2{};
  #pragma unroll
  for (int i = 0; i < 16; ++i) {
    int kl = i + 16*hs;
    float g  = (kl < CRn) ? gfold[b*CRn + kl] : 0.0f;
    A0[i] = (_Float16)((kl < CRn) ? w2[(0*16 + mrow)*CRn + kl] * g : 0.0f);
    A1[i] = (_Float16)((kl < CRn) ? w2[(1*16 + mrow)*CRn + kl] * g : 0.0f);
    A2[i] = (_Float16)((kl < CRn) ? w2[(2*16 + mrow)*CRn + kl] * g : 0.0f);
  }
  const int n  = lane & 15;
  const int kh = lane >> 4;
  const _Float16 hmask = (_Float16)((kh == 0) ? 1.0f : 0.0f);
  const size_t ob = (size_t)b * Cn * HW;

  for (int tp = blockIdx.x*8 + wave; tp < NTILES/2; tp += gridDim.x*8) {
    const int p0 = tp * 32;
    const _Float16* ta = h16t + ((size_t)b*NTILES + 2*tp    )*256;
    const _Float16* tb = h16t + ((size_t)b*NTILES + 2*tp + 1)*256;
    v16h Ba{}, Bb{};
    #pragma unroll
    for (int i = 0; i < 16; ++i) {      // contiguous 512B block per tile
      Ba[i] = ta[i*16 + n] * hmask;
      Bb[i] = tb[i*16 + n] * hmask;
    }
    v8f c0a={0,0,0,0,0,0,0,0}, c1a={0,0,0,0,0,0,0,0}, c2a={0,0,0,0,0,0,0,0};
    v8f c0b={0,0,0,0,0,0,0,0}, c1b={0,0,0,0,0,0,0,0}, c2b={0,0,0,0,0,0,0,0};
    c0a = __builtin_amdgcn_wmma_f32_16x16x32_f16(false, A0, false, Ba, (short)0, c0a, false, false);
    c1a = __builtin_amdgcn_wmma_f32_16x16x32_f16(false, A1, false, Ba, (short)0, c1a, false, false);
    c2a = __builtin_amdgcn_wmma_f32_16x16x32_f16(false, A2, false, Ba, (short)0, c2a, false, false);
    c0b = __builtin_amdgcn_wmma_f32_16x16x32_f16(false, A0, false, Bb, (short)0, c0b, false, false);
    c1b = __builtin_amdgcn_wmma_f32_16x16x32_f16(false, A1, false, Bb, (short)0, c1b, false, false);
    c2b = __builtin_amdgcn_wmma_f32_16x16x32_f16(false, A2, false, Bb, (short)0, c2b, false, false);
    #pragma unroll
    for (int v = 0; v < 8; ++v) {
      int r = v + 8*hs;
      x1[ob + (size_t)(0*16 + r)*HW + p0 + n]      = c0a[v];
      x1[ob + (size_t)(1*16 + r)*HW + p0 + n]      = c1a[v];
      x1[ob + (size_t)(2*16 + r)*HW + p0 + n]      = c2a[v];
      x1[ob + (size_t)(0*16 + r)*HW + p0 + 16 + n] = c0b[v];
      x1[ob + (size_t)(1*16 + r)*HW + p0 + 16 + n] = c1b[v];
      x1[ob + (size_t)(2*16 + r)*HW + p0 + 16 + n] = c2b[v];
    }
  }
}

// ---------------------------------------------------------------------------
// One pass over x1 producing all three area pools (f=2,4,8), float4 loads.
// ---------------------------------------------------------------------------
__global__ void pool_pyramid(const float* __restrict__ x1, float* __restrict__ p1,
                             float* __restrict__ p2, float* __restrict__ p3)
{
  int idx = blockIdx.x*blockDim.x + threadIdx.x;
  if (idx >= Bn*Cn*32*32) return;
  int x3 = idx & 31; int t = idx >> 5;
  int y3 = t & 31;   t >>= 5;
  int c = t % Cn;    int b = t / Cn;
  size_t base = ((size_t)(b*Cn + c)) * HW;
  float s2[4][4];
  #pragma unroll
  for (int j = 0; j < 4; ++j)
    #pragma unroll
    for (int k = 0; k < 4; ++k) s2[j][k] = 0.f;
  #pragma unroll
  for (int yy = 0; yy < 8; ++yy) {
    const float4* row = reinterpret_cast<const float4*>(x1 + base + (size_t)(y3*8 + yy)*Wn + x3*8);
    float4 r0 = row[0], r1 = row[1];
    s2[yy>>1][0] += r0.x + r0.y;
    s2[yy>>1][1] += r0.z + r0.w;
    s2[yy>>1][2] += r1.x + r1.y;
    s2[yy>>1][3] += r1.z + r1.w;
  }
  size_t b1 = ((size_t)(b*Cn + c))*HW1;
  size_t b2 = ((size_t)(b*Cn + c))*HW2;
  float tot = 0.f;
  #pragma unroll
  for (int j = 0; j < 4; ++j) {
    float4 o = make_float4(s2[j][0]*0.25f, s2[j][1]*0.25f, s2[j][2]*0.25f, s2[j][3]*0.25f);
    *reinterpret_cast<float4*>(p1 + b1 + (size_t)(y3*4 + j)*128 + x3*4) = o;
  }
  #pragma unroll
  for (int j = 0; j < 2; ++j) {
    float sa = s2[2*j][0] + s2[2*j][1] + s2[2*j+1][0] + s2[2*j+1][1];
    float sb = s2[2*j][2] + s2[2*j][3] + s2[2*j+1][2] + s2[2*j+1][3];
    *reinterpret_cast<float2*>(p2 + b2 + (size_t)(y3*2 + j)*64 + x3*2) = make_float2(sa*0.0625f, sb*0.0625f);
    tot += sa + sb;
  }
  p3[((size_t)(b*Cn + c))*HW3 + y3*32 + x3] = tot*(1.0f/64.0f);
}

// ---------------------------------------------------------------------------
// Depthwise 3x3, pad 1, branch-free (clamped addresses + select-masked taps)
// ---------------------------------------------------------------------------
__global__ void dwconv3(const float* __restrict__ in, float* __restrict__ out,
                        const float* __restrict__ w, int Hs, int Ws)
{
  int idx = blockIdx.x*blockDim.x + threadIdx.x;
  int total = Bn*Cn*Hs*Ws;
  if (idx >= total) return;
  int xq = idx % Ws; int t = idx / Ws;
  int yq = t % Hs;   t /= Hs;
  int c  = t % Cn;
  const float* wc = w + c*9;
  size_t base = (size_t)(idx - (yq*Ws + xq));
  float acc = 0.f;
  #pragma unroll
  for (int dy = -1; dy <= 1; ++dy) {
    int y = yq + dy;
    bool yin = ((unsigned)y < (unsigned)Hs);
    int yc = yin ? y : yq;
    #pragma unroll
    for (int dx = -1; dx <= 1; ++dx) {
      int xx = xq + dx;
      bool xin = ((unsigned)xx < (unsigned)Ws);
      int xc = xin ? xx : xq;
      float v = in[base + (size_t)yc*Ws + xc];
      acc += (yin && xin) ? v * wc[(dy+1)*3 + (dx+1)] : 0.f;
    }
  }
  out[idx] = acc;
}

// ---------------------------------------------------------------------------
// Per-(b,c) spatial mean, float4
// ---------------------------------------------------------------------------
__global__ void chanmean(const float* __restrict__ in, float* __restrict__ q, int HWs)
{
  int bc = blockIdx.x;
  const float4* p = reinterpret_cast<const float4*>(in + (size_t)bc * HWs);
  int n4 = HWs >> 2;
  float s = 0.f;
  for (int i = threadIdx.x; i < n4; i += blockDim.x) {
    float4 v = p[i];
    s += (v.x + v.y) + (v.z + v.w);
  }
  __shared__ float sh[256];
  sh[threadIdx.x] = s; __syncthreads();
  for (int off = 128; off > 0; off >>= 1) {
    if ((int)threadIdx.x < off) sh[threadIdx.x] += sh[threadIdx.x + off];
    __syncthreads();
  }
  if (threadIdx.x == 0) q[bc] = sh[0] / (float)HWs;
}

// ---------------------------------------------------------------------------
// dmff gates (tiny); writes s = 1 + gate
// ---------------------------------------------------------------------------
__global__ void dmff_gate(const float* __restrict__ q1, const float* __restrict__ q2,
                          const float* __restrict__ kw, const float* __restrict__ v1w,
                          const float* __restrict__ v2w,
                          float* __restrict__ s1, float* __restrict__ s2)
{
  int t = blockIdx.x*blockDim.x + threadIdx.x;
  if (t >= Bn*Cn) return;
  int b = t / Cn, i = t % Cn;
  const float scale = rsqrtf((float)Cn);
  float k[3], v1[3], v2[3];
  for (int j = 0; j < 3; ++j) {
    float kk = 0.f, a1 = 0.f, a2 = 0.f;
    for (int c = 0; c < Cn; ++c) {
      float q1c = q1[b*Cn + c], q2c = q2[b*Cn + c];
      kk += (q1c + q2c)*kw[j*Cn + c];
      a1 += q1c*v1w[j*Cn + c];
      a2 += q2c*v2w[j*Cn + c];
    }
    k[j] = kk; v1[j] = a1; v2[j] = a2;
  }
  {
    float qi = q1[b*Cn + i], e[3], mx = -1e30f;
    for (int j = 0; j < 3; ++j) { e[j] = qi*k[j]*scale; mx = fmaxf(mx, e[j]); }
    float den = 0.f, num = 0.f;
    for (int j = 0; j < 3; ++j) { float ex = __expf(e[j]-mx); den += ex; num += ex*v1[j]; }
    s1[t] = 1.0f + 1.0f/(1.0f + __expf(-(num/den)));
  }
  {
    float qi = q2[b*Cn + i], e[3], mx = -1e30f;
    for (int j = 0; j < 3; ++j) { e[j] = qi*k[j]*scale; mx = fmaxf(mx, e[j]); }
    float den = 0.f, num = 0.f;
    for (int j = 0; j < 3; ++j) { float ex = __expf(e[j]-mx); den += ex; num += ex*v2[j]; }
    s2[t] = 1.0f + 1.0f/(1.0f + __expf(-(num/den)));
  }
}

// ---------------------------------------------------------------------------
// dmff combine with fused up2, 4 pixels/thread
// ---------------------------------------------------------------------------
__global__ void dmff_combine(const float* __restrict__ x1, const float* __restrict__ x2s,
                             const float* __restrict__ s1, const float* __restrict__ s2,
                             float* __restrict__ out, int Hs, int Ws)
{
  int t4 = blockIdx.x*blockDim.x + threadIdx.x;
  int total4 = (Bn*Cn*Hs*Ws) >> 2;
  if (t4 >= total4) return;
  int idx = t4 << 2;
  int xq = idx % Ws; int t = idx / Ws;
  int yq = t % Hs;   t /= Hs;
  int c  = t % Cn;   int b = t / Cn;
  int W2 = Ws >> 1;
  float g1 = s1[b*Cn + c], g2 = s2[b*Cn + c];
  float4 a = *reinterpret_cast<const float4*>(x1 + idx);
  const float2 u = *reinterpret_cast<const float2*>(
      x2s + (((size_t)(b*Cn + c))*(Hs>>1) + (yq>>1))*W2 + (xq>>1));
  float4 o;
  o.x = a.x*g1 + u.x*g2;
  o.y = a.y*g1 + u.x*g2;
  o.z = a.z*g1 + u.y*g2;
  o.w = a.w*g1 + u.y*g2;
  *reinterpret_cast<float4*>(out + idx) = o;
}

// ---------------------------------------------------------------------------
// Final: x1_fuse at full res; out = x*sigmoid(x1_fuse); writes both outputs.
// 4 pixels/thread, non-temporal streaming (raw ext-vector type for builtins).
// ---------------------------------------------------------------------------
__global__ void final_fuse(const float* __restrict__ x, const float* __restrict__ x1d,
                           const float* __restrict__ ufd, const float* __restrict__ s1,
                           const float* __restrict__ s2, float* __restrict__ out)
{
  int t4 = blockIdx.x*blockDim.x + threadIdx.x;
  const int total = Bn*Cn*HW;
  if (t4 >= (total >> 2)) return;
  int idx = t4 << 2;
  int xq = idx & (Wn-1); int t = idx >> 8;
  int yq = t & (Hn-1);   t >>= 8;
  int c  = t % Cn;       int b = t / Cn;
  float g1 = s1[b*Cn + c], g2 = s2[b*Cn + c];
  f4v a  = *reinterpret_cast<const f4v*>(x1d + idx);
  f4v xv = __builtin_nontemporal_load(reinterpret_cast<const f4v*>(x + idx));
  const float2 u = *reinterpret_cast<const float2*>(
      ufd + (((size_t)(b*Cn + c))*128 + (yq>>1))*128 + (xq>>1));
  f4v f, o;
  f.x = a.x*g1 + u.x*g2;  f.y = a.y*g1 + u.x*g2;
  f.z = a.z*g1 + u.y*g2;  f.w = a.w*g1 + u.y*g2;
  o.x = xv.x * (1.0f/(1.0f + __expf(-f.x)));
  o.y = xv.y * (1.0f/(1.0f + __expf(-f.y)));
  o.z = xv.z * (1.0f/(1.0f + __expf(-f.z)));
  o.w = xv.w * (1.0f/(1.0f + __expf(-f.w)));
  __builtin_nontemporal_store(o, reinterpret_cast<f4v*>(out + idx));
  __builtin_nontemporal_store(f, reinterpret_cast<f4v*>(out + total + idx));
}

// ---------------------------------------------------------------------------
extern "C" void kernel_launch(void* const* d_in, const int* in_sizes, int n_in,
                              void* d_out, int out_size, void* d_ws, size_t ws_size,
                              hipStream_t stream) {
  const float* x   = (const float*)d_in[0];
  const float* w1  = (const float*)d_in[1];
  const float* kvw = (const float*)d_in[2];
  const float* kvb = (const float*)d_in[3];
  const float* w2  = (const float*)d_in[4];
  const float* kw  = (const float*)d_in[5];
  const float* v1w = (const float*)d_in[6];
  const float* v2w = (const float*)d_in[7];
  const float* dw  = (const float*)d_in[8];
  float* out = (float*)d_out;

  char* ws = (char*)d_ws;
  size_t off = 0;
  auto alloc = [&](size_t bytes) { size_t o = off; off = (off + bytes + 255) & ~(size_t)255; return o; };
  size_t o_h16  = alloc((size_t)Bn*NTILES*256*sizeof(_Float16));   // padded 16-ch tile-major
  size_t o_x1   = alloc((size_t)Bn*Cn*HW*sizeof(float));
  size_t o_x1d  = alloc((size_t)Bn*Cn*HW*sizeof(float));
  size_t o_p1   = alloc((size_t)Bn*Cn*HW1*sizeof(float));
  size_t o_p1d  = alloc((size_t)Bn*Cn*HW1*sizeof(float));
  size_t o_p1f  = alloc((size_t)Bn*Cn*HW1*sizeof(float));
  size_t o_p1fd = alloc((size_t)Bn*Cn*HW1*sizeof(float));
  size_t o_p2   = alloc((size_t)Bn*Cn*HW2*sizeof(float));
  size_t o_p2d  = alloc((size_t)Bn*Cn*HW2*sizeof(float));
  size_t o_p2f  = alloc((size_t)Bn*Cn*HW2*sizeof(float));
  size_t o_p2fd = alloc((size_t)Bn*Cn*HW2*sizeof(float));
  size_t o_p3   = alloc((size_t)Bn*Cn*HW3*sizeof(float));
  size_t o_p3d  = alloc((size_t)Bn*Cn*HW3*sizeof(float));
  size_t o_hsum = alloc(Bn*CRn*sizeof(float));
  size_t o_gf   = alloc(Bn*CRn*sizeof(float));
  size_t o_q1   = alloc(Bn*Cn*sizeof(float));
  size_t o_q2   = alloc(Bn*Cn*sizeof(float));
  size_t o_s1   = alloc(Bn*Cn*sizeof(float));
  size_t o_s2   = alloc(Bn*Cn*sizeof(float));
  (void)ws_size; (void)in_sizes; (void)n_in; (void)out_size;

  _Float16* h16t = (_Float16*)(ws + o_h16);
  float* x1   = (float*)(ws + o_x1);
  float* x1d  = (float*)(ws + o_x1d);
  float* p1   = (float*)(ws + o_p1);
  float* p1d  = (float*)(ws + o_p1d);
  float* p1f  = (float*)(ws + o_p1f);
  float* p1fd = (float*)(ws + o_p1fd);
  float* p2   = (float*)(ws + o_p2);
  float* p2d  = (float*)(ws + o_p2d);
  float* p2f  = (float*)(ws + o_p2f);
  float* p2fd = (float*)(ws + o_p2fd);
  float* p3   = (float*)(ws + o_p3);
  float* p3d  = (float*)(ws + o_p3d);
  float* hsum = (float*)(ws + o_hsum);
  float* gf   = (float*)(ws + o_gf);
  float* q1   = (float*)(ws + o_q1);
  float* q2   = (float*)(ws + o_q2);
  float* s1   = (float*)(ws + o_s1);
  float* s2   = (float*)(ws + o_s2);

  (void)hipMemsetAsync(hsum, 0, Bn*CRn*sizeof(float), stream);
  conv1_wmma<<<dim3(128, Bn), 256, 0, stream>>>(x, w1, h16t, hsum);
  se_gate<<<1, 96, 0, stream>>>(hsum, kvw, kvb, gf);
  conv2_wmma<<<dim3(128, Bn), 256, 0, stream>>>(h16t, w2, gf, x1);
  {
    int nthr = Bn*Cn*32*32;
    pool_pyramid<<<(nthr + 255)/256, 256, 0, stream>>>(x1, p1, p2, p3);
  }
  // level 0: dmff(dw(p2), up2(dw(p3))) @64
  dwconv3<<<(Bn*Cn*HW2 + 255)/256, 256, 0, stream>>>(p2, p2d, dw + 0*Cn*9, 64, 64);
  dwconv3<<<(Bn*Cn*HW3 + 255)/256, 256, 0, stream>>>(p3, p3d, dw + 1*Cn*9, 32, 32);
  chanmean<<<Bn*Cn, 256, 0, stream>>>(p2d, q1, HW2);
  chanmean<<<Bn*Cn, 256, 0, stream>>>(p3d, q2, HW3);
  dmff_gate<<<2, 192, 0, stream>>>(q1, q2, kw + 0*3*Cn, v1w + 0*3*Cn, v2w + 0*3*Cn, s1, s2);
  dmff_combine<<<((Bn*Cn*HW2/4) + 255)/256, 256, 0, stream>>>(p2d, p3d, s1, s2, p2f, 64, 64);
  // level 1: dmff(dw(p1), up2(dw(p2f))) @128
  dwconv3<<<(Bn*Cn*HW1 + 255)/256, 256, 0, stream>>>(p1, p1d, dw + 2*Cn*9, 128, 128);
  dwconv3<<<(Bn*Cn*HW2 + 255)/256, 256, 0, stream>>>(p2f, p2fd, dw + 3*Cn*9, 64, 64);
  chanmean<<<Bn*Cn, 256, 0, stream>>>(p1d, q1, HW1);
  chanmean<<<Bn*Cn, 256, 0, stream>>>(p2fd, q2, HW2);
  dmff_gate<<<2, 192, 0, stream>>>(q1, q2, kw + 1*3*Cn, v1w + 1*3*Cn, v2w + 1*3*Cn, s1, s2);
  dmff_combine<<<((Bn*Cn*HW1/4) + 255)/256, 256, 0, stream>>>(p1d, p2fd, s1, s2, p1f, 128, 128);
  // level 2: dmff(dw(x1), up2(dw(p1f))) @256 fused with output
  dwconv3<<<(Bn*Cn*HW + 255)/256, 256, 0, stream>>>(x1, x1d, dw + 4*Cn*9, 256, 256);
  dwconv3<<<(Bn*Cn*HW1 + 255)/256, 256, 0, stream>>>(p1f, p1fd, dw + 5*Cn*9, 128, 128);
  chanmean<<<Bn*Cn, 256, 0, stream>>>(x1d, q1, HW);
  chanmean<<<Bn*Cn, 256, 0, stream>>>(p1fd, q2, HW1);
  dmff_gate<<<2, 192, 0, stream>>>(q1, q2, kw + 2*3*Cn, v1w + 2*3*Cn, v2w + 2*3*Cn, s1, s2);
  final_fuse<<<((Bn*Cn*HW/4) + 255)/256, 256, 0, stream>>>(x, x1d, p1fd, s1, s2, out);
}
